// LinearAttention_9002251452759
// MI455X (gfx1250) — matrix operand order repaired and checked by reading, not compiled, not run on hardware
//
#include <hip/hip_runtime.h>
#include <cstdint>

// ---------------- problem constants ----------------
static constexpr int Bb    = 2;
static constexpr int Nn    = 8192;
static constexpr int Dd    = 1024;
static constexpr int Hh    = 8;
static constexpr int DH    = 64;
static constexpr int INNER = Hh * DH;       // 512
static constexpr int MROWS = Bb * Nn;       // 16384
static constexpr int QKVC  = 3 * INNER;     // 1536

// ---------------- vector / WMMA types ----------------
typedef __attribute__((ext_vector_type(16))) __bf16   v16bf;
typedef __attribute__((ext_vector_type(8)))  float    v8f;
typedef __attribute__((ext_vector_type(4)))  unsigned u32x4;
typedef __attribute__((ext_vector_type(4)))  float    f32x4;

union ABFrag  { v16bf v; u32x4 q[2]; uint32_t u[8]; uint16_t h[16]; };
union AccFrag { v8f v; float f[8]; };
union H8      { u32x4 q; uint16_t h[8]; };
union H16     { u32x4 q[2]; uint16_t h[16]; };
union F16     { f32x4 v[4]; float f[16]; };

__device__ __forceinline__ uint16_t f32_to_bf16(float f) {
  union { float f; uint32_t u; } x; x.f = f;
  uint32_t r = x.u + 0x7FFFu + ((x.u >> 16) & 1u);   // round-to-nearest-even
  return (uint16_t)(r >> 16);
}

__device__ __forceinline__ v8f wmma_bf16(const ABFrag& a, const ABFrag& b, v8f c) {
  return __builtin_amdgcn_wmma_f32_16x16x32_bf16(false, a.v, false, b.v,
                                                 (short)0, c, false, false);
}

// =====================================================================
// fp32 [R][C] -> bf16 transposed [C][R]  (one-time weight prep)
// =====================================================================
__global__ __launch_bounds__(256) void cvt_transpose_kernel(
    const float* __restrict__ in, uint16_t* __restrict__ out, int R, int C) {
  int i = blockIdx.x * 256 + threadIdx.x;
  if (i < R * C) {
    int r = i / C, c = i - r * C;
    out[(size_t)c * R + r] = f32_to_bf16(in[i]);
  }
}

__global__ __launch_bounds__(256) void zero_f32_kernel(float* __restrict__ p, int n) {
  int i = blockIdx.x * 256 + threadIdx.x;
  if (i < n) p[i] = 0.0f;
}

// =====================================================================
// Tiled bf16 WMMA GEMM:  C[M,N] = A[M,K] * B[K,N] (+ bias)
// B supplied TRANSPOSED (BwT[N][K] bf16) so all staging is contiguous b128.
// Block tile 128x128, K-step 32, 8 waves (4M x 2N), 8 WMMA / wave / K-step.
// Double-buffered LDS: next tile loads overlap current WMMAs; 1 barrier/iter.
// =====================================================================
template<bool A_IS_F32, bool ADD_BIAS>
__global__ __launch_bounds__(256) void gemm_wmma_kernel(
    const void* __restrict__ Ap, const uint16_t* __restrict__ BwT,
    float* __restrict__ Cp, const float* __restrict__ bias,
    int M, int K, int Ncols)
{
  __shared__ alignas(16) uint16_t As[2][128 * 32];   // [m][k]
  __shared__ alignas(16) uint16_t Bt[2][128 * 32];   // [n][k]

  const int tid   = threadIdx.x;
  const int wave  = tid >> 5;
  const int lane  = tid & 31;
  const int waveM = wave >> 1;         // 0..3  -> 32-row strip
  const int waveN = wave & 1;          // 0..1  -> 64-col strip
  const int m0    = blockIdx.y * 128;
  const int n0    = blockIdx.x * 128;

  const int r  = tid >> 1;             // 0..127 : A row / B column
  const int cs = (tid & 1) * 16;       // 0 / 16 : k segment

  const float*    Af = (const float*)Ap;
  const uint16_t* Ah = (const uint16_t*)Ap;
  const uint16_t* bsrc = BwT + (size_t)(n0 + r) * K + cs;

  AccFrag acc[2][4];
#pragma unroll
  for (int mi = 0; mi < 2; ++mi)
#pragma unroll
    for (int t = 0; t < 4; ++t)
#pragma unroll
      for (int j = 0; j < 8; ++j) acc[mi][t].f[j] = 0.0f;

  const int mlo = lane & 15;
  const int khf = lane >> 4;

  // ---- prologue: stage K-tile 0 into buffer 0 ----
  {
    uint16_t* dA = &As[0][r * 32 + cs];
    if (A_IS_F32) {
      const float* s = Af + (size_t)(m0 + r) * K + cs;
      H16 w;
#pragma unroll
      for (int j = 0; j < 16; ++j) w.h[j] = f32_to_bf16(s[j]);
      ((u32x4*)dA)[0] = w.q[0]; ((u32x4*)dA)[1] = w.q[1];
    } else {
      const uint16_t* s = Ah + (size_t)(m0 + r) * K + cs;
      ((u32x4*)dA)[0] = ((const u32x4*)s)[0];
      ((u32x4*)dA)[1] = ((const u32x4*)s)[1];
    }
    uint16_t* dB = &Bt[0][r * 32 + cs];
    ((u32x4*)dB)[0] = ((const u32x4*)bsrc)[0];
    ((u32x4*)dB)[1] = ((const u32x4*)bsrc)[1];
  }
  __syncthreads();

  const int KT = K >> 5;
  for (int kt = 0; kt < KT; ++kt) {
    const int  cur  = kt & 1;
    const bool have = (kt + 1) < KT;

    // ---- issue next tile's global loads into registers ----
    F16 fa; H16 wa, wb;
    if (have) {
      const int k0n = (kt + 1) << 5;
      if (A_IS_F32) {
        const f32x4* s = (const f32x4*)(Af + (size_t)(m0 + r) * K + k0n + cs);
#pragma unroll
        for (int j = 0; j < 4; ++j) fa.v[j] = s[j];
      } else {
        const uint16_t* s = Ah + (size_t)(m0 + r) * K + k0n + cs;
        wa.q[0] = ((const u32x4*)s)[0];
        wa.q[1] = ((const u32x4*)s)[1];
      }
      const uint16_t* s2 = bsrc + k0n;
      wb.q[0] = ((const u32x4*)s2)[0];
      wb.q[1] = ((const u32x4*)s2)[1];
    }

    // ---- compute on current buffer: 12 ds_load_b128 + 8 WMMA ----
    ABFrag a[2];
#pragma unroll
    for (int mi = 0; mi < 2; ++mi) {
      const uint16_t* base = &As[cur][(waveM * 32 + mi * 16 + mlo) * 32];
      a[mi].q[0] = *(const u32x4*)(base + 8 * khf);
      a[mi].q[1] = *(const u32x4*)(base + 16 + 8 * khf);
    }
#pragma unroll
    for (int t = 0; t < 4; ++t) {
      ABFrag b;
      const uint16_t* base = &Bt[cur][(waveN * 64 + t * 16 + mlo) * 32 + 8 * khf];
      b.q[0] = *(const u32x4*)(base);
      b.q[1] = *(const u32x4*)(base + 16);
      acc[0][t].v = wmma_bf16(a[0], b, acc[0][t].v);
      acc[1][t].v = wmma_bf16(a[1], b, acc[1][t].v);
    }

    // ---- commit next tile to the other buffer ----
    if (have) {
      if (A_IS_F32) {
#pragma unroll
        for (int j = 0; j < 16; ++j) wa.h[j] = f32_to_bf16(fa.f[j]);
      }
      uint16_t* dA = &As[cur ^ 1][r * 32 + cs];
      ((u32x4*)dA)[0] = wa.q[0]; ((u32x4*)dA)[1] = wa.q[1];
      uint16_t* dB = &Bt[cur ^ 1][r * 32 + cs];
      ((u32x4*)dB)[0] = wb.q[0]; ((u32x4*)dB)[1] = wb.q[1];
    }
    __syncthreads();
  }

  // ---- epilogue: C layout (lane hi-half -> M+8) ----
  const int mh = lane >> 4;
#pragma unroll
  for (int mi = 0; mi < 2; ++mi) {
#pragma unroll
    for (int t = 0; t < 4; ++t) {
      const int gn = n0 + waveN * 64 + t * 16 + mlo;
      const float bv = ADD_BIAS ? bias[gn] : 0.0f;
#pragma unroll
      for (int r8 = 0; r8 < 8; ++r8) {
        const int gm = m0 + waveM * 32 + mi * 16 + r8 + 8 * mh;
        Cp[(size_t)gm * Ncols + gn] = acc[mi][t].f[r8] + bv;
      }
    }
  }
}

// =====================================================================
// softmax over DH=64 for q (one wave per (b,h,n) row), fused with v->bf16.
// q written [bh][n][d] (rows feed attn_out A-frags);
// v written TRANSPOSED [bh][d][n] (columns feed context B-frags).
// =====================================================================
__global__ __launch_bounds__(256) void softmax_q_cvt_v_kernel(
    const float* __restrict__ qkv, uint16_t* __restrict__ qsm,
    uint16_t* __restrict__ vT)
{
  const int wrow = blockIdx.x * 8 + (threadIdx.x >> 5);   // (b*H+h)*N + n
  const int lane = threadIdx.x & 31;
  const int n  = wrow & (Nn - 1);
  const int bh = wrow >> 13;
  const int h  = bh & (Hh - 1);
  const int b  = bh >> 3;

  const float* base = qkv + (size_t)(b * Nn + n) * QKVC + h * DH;  // q slice
  const float* vrow = base + 2 * INNER;                            // v slice

  float x0 = base[lane * 2], x1 = base[lane * 2 + 1];
  float mx = fmaxf(x0, x1);
#pragma unroll
  for (int s = 16; s > 0; s >>= 1) mx = fmaxf(mx, __shfl_xor(mx, s, 32));
  float e0 = __expf(x0 - mx), e1 = __expf(x1 - mx);
  float sum = e0 + e1;
#pragma unroll
  for (int s = 16; s > 0; s >>= 1) sum += __shfl_xor(sum, s, 32);
  const float inv = 1.0f / sum;

  uint16_t* qd = qsm + (size_t)wrow * DH;
  qd[lane * 2]     = f32_to_bf16(e0 * inv);
  qd[lane * 2 + 1] = f32_to_bf16(e1 * inv);

  uint16_t* vd = vT + ((size_t)bh * DH) * Nn + n;          // [bh][d][n]
  vd[(size_t)(lane * 2) * Nn]     = f32_to_bf16(vrow[lane * 2]);
  vd[(size_t)(lane * 2 + 1) * Nn] = f32_to_bf16(vrow[lane * 2 + 1]);
}

// =====================================================================
// softmax over N=8192 for k. One 1024-thread block per (b,h).
// Lane owns d = tid&63 (coalesced 256B bursts); 16 n-strips reduced in LDS.
// Output TRANSPOSED [bh][d][n] (columns feed context A-frags).
// =====================================================================
__global__ __launch_bounds__(1024) void softmax_k_kernel(
    const float* __restrict__ qkv, uint16_t* __restrict__ kT)
{
  __shared__ float redm[16][64];
  __shared__ float reds[16][64];
  const int bh = blockIdx.x;
  const int b = bh >> 3, h = bh & (Hh - 1);
  const int tid = threadIdx.x;
  const int d = tid & 63;
  const int strip = tid >> 6;               // 0..15

  const float* base = qkv + (size_t)b * Nn * QKVC + INNER + h * DH + d;

  float mx = -3.4e38f;
  for (int n = strip; n < Nn; n += 16) mx = fmaxf(mx, base[(size_t)n * QKVC]);
  redm[strip][d] = mx; __syncthreads();
  if (strip == 0) {
    float m2 = redm[0][d];
#pragma unroll
    for (int s = 1; s < 16; ++s) m2 = fmaxf(m2, redm[s][d]);
    redm[0][d] = m2;
  }
  __syncthreads();
  mx = redm[0][d];

  float sum = 0.0f;
  for (int n = strip; n < Nn; n += 16) sum += __expf(base[(size_t)n * QKVC] - mx);
  reds[strip][d] = sum; __syncthreads();
  if (strip == 0) {
    float s2 = 0.0f;
#pragma unroll
    for (int s = 0; s < 16; ++s) s2 += reds[s][d];
    reds[0][d] = s2;
  }
  __syncthreads();
  const float inv = 1.0f / reds[0][d];

  uint16_t* out = kT + ((size_t)bh * DH + d) * Nn;          // [bh][d][n]
  for (int n = strip; n < Nn; n += 16)
    out[n] = f32_to_bf16(__expf(base[(size_t)n * QKVC] - mx) * inv);
}

// =====================================================================
// context[b,h] += k^T v over an n-chunk of 1024.
// Inputs already [d][n] / [e][n] so staging is pure contiguous b128.
// Double-buffered LDS, one barrier per 32-step; atomic f32 accumulate.
// =====================================================================
__global__ __launch_bounds__(512) void context_kernel(
    const uint16_t* __restrict__ kT, const uint16_t* __restrict__ vT,
    float* __restrict__ ctx)
{
  __shared__ alignas(16) uint16_t kS[2][64 * 32];   // [d][n]
  __shared__ alignas(16) uint16_t vS[2][64 * 32];   // [e][n]
  const int bh    = blockIdx.x >> 3;
  const int split = blockIdx.x & 7;
  const int tid   = threadIdx.x;
  const int wave  = tid >> 5, lane = tid & 31;
  const int dm = (wave >> 2) * 16;     // A row tile (d)
  const int de = (wave & 3) * 16;      // B col tile (e)

  const int mlo = lane & 15;
  const int khf = lane >> 4;

  // staging role: threads 0..255 -> k, 256..511 -> v; one b128 each
  const int half = tid >> 8;
  const int st   = tid & 255;
  const int drow = st >> 2;            // 0..63
  const int nseg = (st & 3) * 8;       // 0,8,16,24
  const uint16_t* gsrc =
      (half ? vT : kT) + ((size_t)bh * DH + drow) * Nn + nseg;
  uint16_t* dsty = (half ? vS[0] : kS[0]);
  uint16_t* dstn = (half ? vS[1] : kS[1]);
  const int doff = drow * 32 + nseg;

  AccFrag acc;
#pragma unroll
  for (int j = 0; j < 8; ++j) acc.f[j] = 0.0f;

  const int nstart = split * (Nn / 8);
  // prologue: stage chunk 0 into buffer 0
  *(u32x4*)(dsty + doff) = *(const u32x4*)(gsrc + nstart);
  __syncthreads();

  const int NSTEPS = (Nn / 8) / 32;    // 32
  for (int it = 0; it < NSTEPS; ++it) {
    const int  cur  = it & 1;
    const bool have = (it + 1) < NSTEPS;

    H8 w;
    if (have) w.q = *(const u32x4*)(gsrc + nstart + (it + 1) * 32);

    ABFrag a, bfr;
    {
      const uint16_t* base = (cur ? kS[1] : kS[0]) + (dm + mlo) * 32;
      a.q[0] = *(const u32x4*)(base + 8 * khf);
      a.q[1] = *(const u32x4*)(base + 16 + 8 * khf);
    }
    {
      const uint16_t* base = (cur ? vS[1] : vS[0]) + (de + mlo) * 32 + 8 * khf;
      bfr.q[0] = *(const u32x4*)(base);
      bfr.q[1] = *(const u32x4*)(base + 16);
    }
    acc.v = wmma_bf16(a, bfr, acc.v);

    if (have) *(u32x4*)((cur ? dsty : dstn) + doff) = w.q;
    __syncthreads();
  }

  const int mh = lane >> 4;
  float* cb = ctx + (size_t)bh * DH * DH;
#pragma unroll
  for (int r = 0; r < 8; ++r)
    atomicAdd(&cb[(size_t)(dm + r + 8 * mh) * DH + de + mlo], acc.f[r]);
}

// =====================================================================
// out_h = q_sm @ context per (b,h). Context transposed to LDS bf16;
// A-fragments are direct global b128 loads from contiguous q rows.
// =====================================================================
__global__ __launch_bounds__(256) void attn_out_kernel(
    const uint16_t* __restrict__ qsm, const float* __restrict__ ctx,
    uint16_t* __restrict__ attn)
{
  __shared__ alignas(16) uint16_t ctxT[DH * DH];       // [e][d]
  const int bh = blockIdx.x >> 6;
  const int rb = blockIdx.x & 63;
  const int b = bh >> 3, h = bh & (Hh - 1);
  const int tid = threadIdx.x;

  {
    const float* cb = ctx + (size_t)bh * DH * DH;
    for (int i = tid; i < DH * DH; i += 256) {
      const int d = i >> 6, e = i & 63;
      ctxT[e * DH + d] = f32_to_bf16(cb[i]);
    }
  }
  __syncthreads();

  const int wave = tid >> 5, lane = tid & 31;
  const int nrow0 = rb * 128 + wave * 16;
  const uint16_t* qb = qsm + ((size_t)bh * Nn + nrow0) * DH;
  const int mlo = lane & 15;
  const int khf = lane >> 4;

  AccFrag acc[4];
#pragma unroll
  for (int t = 0; t < 4; ++t)
#pragma unroll
    for (int j = 0; j < 8; ++j) acc[t].f[j] = 0.0f;

#pragma unroll
  for (int ks = 0; ks < 2; ++ks) {         // K = 64 -> two 32-steps
    ABFrag a;
    const uint16_t* arow = qb + (size_t)mlo * DH + ks * 32;
    a.q[0] = *(const u32x4*)(arow + 8 * khf);
    a.q[1] = *(const u32x4*)(arow + 16 + 8 * khf);
#pragma unroll
    for (int t = 0; t < 4; ++t) {
      ABFrag bfr;
      const uint16_t* bbase = &ctxT[(t * 16 + mlo) * DH + ks * 32 + 8 * khf];
      bfr.q[0] = *(const u32x4*)(bbase);
      bfr.q[1] = *(const u32x4*)(bbase + 16);
      acc[t].v = wmma_bf16(a, bfr, acc[t].v);
    }
  }

  const int mh = lane >> 4;
#pragma unroll
  for (int t = 0; t < 4; ++t) {
#pragma unroll
    for (int r = 0; r < 8; ++r) {
      const int n = nrow0 + r + 8 * mh;
      attn[((size_t)(b * Nn + n)) * INNER + h * DH + t * 16 + mlo] =
          f32_to_bf16(acc[t].f[r]);
    }
  }
}

// =====================================================================
// host-side launcher
// =====================================================================
extern "C" void kernel_launch(void* const* d_in, const int* in_sizes, int n_in,
                              void* d_out, int out_size, void* d_ws, size_t ws_size,
                              hipStream_t stream) {
  (void)in_sizes; (void)n_in; (void)out_size; (void)ws_size;
  const float* x    = (const float*)d_in[0];   // [B, N, D]
  const float* Wqkv = (const float*)d_in[1];   // [D, 3*INNER]
  const float* Wout = (const float*)d_in[2];   // [INNER, D]
  const float* bout = (const float*)d_in[3];   // [D]
  float* out = (float*)d_out;                  // [B, N, D]

  // workspace carve-up
  char* ws = (char*)d_ws;
  float*    qkv      = (float*)ws;    ws += (size_t)MROWS * QKVC * sizeof(float);
  uint16_t* wqkv_bfT = (uint16_t*)ws; ws += (size_t)Dd * QKVC * sizeof(uint16_t);   // [QKVC][Dd]
  uint16_t* wout_bfT = (uint16_t*)ws; ws += (size_t)INNER * Dd * sizeof(uint16_t);  // [Dd][INNER]
  uint16_t* qsm      = (uint16_t*)ws; ws += (size_t)Bb * Hh * Nn * DH * sizeof(uint16_t); // [bh][n][d]
  uint16_t* kT       = (uint16_t*)ws; ws += (size_t)Bb * Hh * Nn * DH * sizeof(uint16_t); // [bh][d][n]
  uint16_t* vT       = (uint16_t*)ws; ws += (size_t)Bb * Hh * Nn * DH * sizeof(uint16_t); // [bh][d][n]
  float*    ctx      = (float*)ws;    ws += (size_t)Bb * Hh * DH * DH * sizeof(float);
  uint16_t* attn     = (uint16_t*)ws; ws += (size_t)MROWS * INNER * sizeof(uint16_t);

  // 1) weights -> bf16 transposed; zero the context accumulator
  cvt_transpose_kernel<<<(Dd * QKVC + 255) / 256, 256, 0, stream>>>(Wqkv, wqkv_bfT, Dd, QKVC);
  cvt_transpose_kernel<<<(INNER * Dd + 255) / 256, 256, 0, stream>>>(Wout, wout_bfT, INNER, Dd);
  zero_f32_kernel<<<(Bb * Hh * DH * DH + 255) / 256, 256, 0, stream>>>(ctx, Bb * Hh * DH * DH);

  // 2) qkv = x @ W_qkv   [16384 x 1024] * [1024 x 1536]
  gemm_wmma_kernel<true, false><<<dim3(QKVC / 128, MROWS / 128), 256, 0, stream>>>(
      x, wqkv_bfT, qkv, nullptr, MROWS, Dd, QKVC);

  // 3) softmax(q over DH) + v->bf16(T) ; softmax(k over N) -> kT
  softmax_q_cvt_v_kernel<<<(Bb * Hh * Nn) / 8, 256, 0, stream>>>(qkv, qsm, vT);
  softmax_k_kernel<<<Bb * Hh, 1024, 0, stream>>>(qkv, kT);

  // 4) context = k^T v per head (8-way split over N, atomic accumulate)
  context_kernel<<<Bb * Hh * 8, 512, 0, stream>>>(kT, vT, ctx);

  // 5) attn = q_sm @ context per head
  attn_out_kernel<<<Bb * Hh * (Nn / 128), 256, 0, stream>>>(qsm, ctx, attn);

  // 6) out = attn @ W_out + b_out   [16384 x 512] * [512 x 1024]
  gemm_wmma_kernel<false, true><<<dim3(Dd / 128, MROWS / 128), 256, 0, stream>>>(
      attn, wout_bfT, out, bout, MROWS, INNER, Dd);
}
